// AttentionLayer_64192581206022
// MI455X (gfx1250) — matrix operand order repaired and checked by reading, not compiled, not run on hardware
//
#include <hip/hip_runtime.h>
#include <math.h>

// ---------------------------------------------------------------------------
// CDNA5 / gfx1250 fused graph-attention pipeline (wave32, WMMA bf16)
// ---------------------------------------------------------------------------

typedef __attribute__((ext_vector_type(8)))  float  v8f;
typedef __attribute__((ext_vector_type(16))) __bf16 v16bf;

#define N_ROWS  8192
#define IN_F    512
#define OUT_F   128
#define LAB_F   64
#define LAB_OUT 64

// f32 -> bf16 (native hardware convert on gfx1250)
__device__ __forceinline__ __bf16 f2bf(float f) { return (__bf16)f; }

// Load one lane's share of a 16x32 A-tile (f32 in memory) and convert to bf16.
// p0 -> 8 contiguous floats (halves j=0..7), p1 -> 8 contiguous floats (j=8..15).
__device__ __forceinline__ v16bf loadA16(const float* __restrict__ p0,
                                         const float* __restrict__ p1) {
  alignas(16) float f[16];
  *(float4*)(f + 0)  = *(const float4*)(p0);
  *(float4*)(f + 4)  = *(const float4*)(p0 + 4);
  *(float4*)(f + 8)  = *(const float4*)(p1);
  *(float4*)(f + 12) = *(const float4*)(p1 + 4);
  v16bf a;
#pragma unroll
  for (int j = 0; j < 16; ++j) a[j] = f2bf(f[j]);
  return a;
}

__device__ __forceinline__ v8f wmma_bf16(v16bf a, v16bf b, v8f c) {
  return __builtin_amdgcn_wmma_f32_16x16x32_bf16(false, a, false, b,
                                                 (short)0, c, false, false);
}

// Swizzled B-operand index for element [k][n] of a KxNW bf16 matrix,
// ncChunks = number of 16-wide column chunks.
__device__ __forceinline__ size_t bIdx(int k, int n, int ncChunks) {
  int kt = k >> 5, kin = k & 31;
  int nc = n >> 4, nl = n & 15;
  int lane = nl + (((kin >> 3) & 1) << 4);
  int j = (kin & 7) + ((kin >> 4) << 3);
  return ((size_t)(kt * ncChunks + nc) * 32 + lane) * 16 + j;
}

// ---------------------------------------------------------------------------
// K1: tmp = y @ Wg  [8192,64], stored bf16 in WMMA-B layout (4 col chunks)
// ---------------------------------------------------------------------------
__global__ __launch_bounds__(256) void k_tmp(const float* __restrict__ y,
                                             const float* __restrict__ Wg,
                                             __bf16* __restrict__ tmpB) {
  int idx = blockIdx.x * 256 + threadIdx.x;      // 8192*64 threads
  int i = idx >> 6, c = idx & 63;
  float s = 0.f;
#pragma unroll
  for (int f = 0; f < LAB_F; ++f) s += y[i * LAB_F + f] * Wg[f * LAB_OUT + c];
  tmpB[bIdx(i, c, 4)] = f2bf(s);
}

// ---------------------------------------------------------------------------
// Pack Wt [512,128] -> bf16 B layout (8 col chunks)
// ---------------------------------------------------------------------------
__global__ __launch_bounds__(256) void k_packWt(const float* __restrict__ Wt,
                                                __bf16* __restrict__ WtB) {
  int idx = blockIdx.x * 256 + threadIdx.x;      // 512*128
  int k = idx >> 7, o = idx & 127;
  WtB[bIdx(k, o, 8)] = f2bf(Wt[idx]);
}

// ---------------------------------------------------------------------------
// K3: xt = x @ Wt via WMMA. Each wave owns a 16-row tile, full K=512.
// ---------------------------------------------------------------------------
__global__ __launch_bounds__(256) void k_xt(const float* __restrict__ x,
                                            const __bf16* __restrict__ WtB,
                                            float* __restrict__ xtF) {
  const int tid = threadIdx.x, w = tid >> 5, t = tid & 31;
  const int m16 = t & 15, g = t >> 4;
  const int row0 = (blockIdx.x * 8 + w) * 16;
  const float* arow = x + (size_t)(row0 + m16) * IN_F;
  v8f acc[8] = {};
  for (int kt = 0; kt < IN_F / 32; ++kt) {
    const int kb = kt * 32;
    v16bf a = loadA16(arow + kb + g * 8, arow + kb + 16 + g * 8);
    const __bf16* bp = WtB + ((size_t)kt * 8 * 32 + t) * 16;
#pragma unroll
    for (int h = 0; h < 2; ++h) {
      v16bf b0 = *(const v16bf*)(bp + (h * 4 + 0) * 512);
      v16bf b1 = *(const v16bf*)(bp + (h * 4 + 1) * 512);
      v16bf b2 = *(const v16bf*)(bp + (h * 4 + 2) * 512);
      v16bf b3 = *(const v16bf*)(bp + (h * 4 + 3) * 512);
      acc[h * 4 + 0] = wmma_bf16(a, b0, acc[h * 4 + 0]);
      acc[h * 4 + 1] = wmma_bf16(a, b1, acc[h * 4 + 1]);
      acc[h * 4 + 2] = wmma_bf16(a, b2, acc[h * 4 + 2]);
      acc[h * 4 + 3] = wmma_bf16(a, b3, acc[h * 4 + 3]);
    }
  }
#pragma unroll
  for (int nc = 0; nc < 8; ++nc)
#pragma unroll
    for (int r = 0; r < 8; ++r)
      xtF[(size_t)(row0 + r + 8 * g) * OUT_F + nc * 16 + m16] = acc[nc][r];
}

// ---------------------------------------------------------------------------
// Pack xt -> bf16 B layout (8 col chunks) for the attention GEMM
// ---------------------------------------------------------------------------
__global__ __launch_bounds__(256) void k_packXt(const float* __restrict__ xtF,
                                                __bf16* __restrict__ xtB) {
  int idx = blockIdx.x * 256 + threadIdx.x;      // 8192*128
  int j = idx >> 7, o = idx & 127;
  xtB[bIdx(j, o, 8)] = f2bf(xtF[idx]);
}

// ---------------------------------------------------------------------------
// K2: y2 = gcn_adj @ tmp. One 16-row tile per block; 8 waves split K=8192,
// deterministic LDS tree reduction of partial C tiles.
// ---------------------------------------------------------------------------
__global__ __launch_bounds__(256) void k_gcn(const float* __restrict__ gcn,
                                             const __bf16* __restrict__ tmpB,
                                             float* __restrict__ y2) {
  __shared__ float s_red[4 * 16 * LAB_OUT];      // 16 KB
  const int tid = threadIdx.x, w = tid >> 5, t = tid & 31;
  const int m16 = t & 15, g = t >> 4;
  const int row0 = blockIdx.x * 16;
  const float* arow = gcn + (size_t)(row0 + m16) * N_ROWS;
  v8f acc[4] = {};
  for (int kt = w; kt < N_ROWS / 32; kt += 8) {
    const int kb = kt * 32;
    __builtin_prefetch(arow + kb + 256, 0, 1);   // next tile for this wave
    v16bf a = loadA16(arow + kb + g * 8, arow + kb + 16 + g * 8);
    const __bf16* bp = tmpB + ((size_t)kt * 4 * 32 + t) * 16;
    v16bf b0 = *(const v16bf*)(bp + 0 * 512);
    v16bf b1 = *(const v16bf*)(bp + 1 * 512);
    v16bf b2 = *(const v16bf*)(bp + 2 * 512);
    v16bf b3 = *(const v16bf*)(bp + 3 * 512);
    acc[0] = wmma_bf16(a, b0, acc[0]);
    acc[1] = wmma_bf16(a, b1, acc[1]);
    acc[2] = wmma_bf16(a, b2, acc[2]);
    acc[3] = wmma_bf16(a, b3, acc[3]);
  }
  for (int step = 4; step >= 1; step >>= 1) {
    if (w >= step && w < 2 * step) {
      float* dst = s_red + (w - step) * (16 * LAB_OUT);
#pragma unroll
      for (int nc = 0; nc < 4; ++nc)
#pragma unroll
        for (int r = 0; r < 8; ++r)
          dst[(r + 8 * g) * LAB_OUT + nc * 16 + m16] = acc[nc][r];
    }
    __syncthreads();
    if (w < step) {
      const float* src = s_red + w * (16 * LAB_OUT);
#pragma unroll
      for (int nc = 0; nc < 4; ++nc)
#pragma unroll
        for (int r = 0; r < 8; ++r)
          acc[nc][r] += src[(r + 8 * g) * LAB_OUT + nc * 16 + m16];
    }
    __syncthreads();
  }
  if (w == 0) {
#pragma unroll
    for (int nc = 0; nc < 4; ++nc)
#pragma unroll
      for (int r = 0; r < 8; ++r)
        y2[(size_t)(row0 + r + 8 * g) * LAB_OUT + nc * 16 + m16] = acc[nc][r];
  }
}

// ---------------------------------------------------------------------------
// u = Wtr @ w1  (64-vector)
// ---------------------------------------------------------------------------
__global__ void k_u(const float* __restrict__ Wtr, const float* __restrict__ w1,
                    float* __restrict__ u) {
  int k = threadIdx.x;   // 64 threads
  float s = 0.f;
#pragma unroll
  for (int o = 0; o < OUT_F; ++o) s += Wtr[k * OUT_F + o] * w1[o];
  u[k] = s;
}

// ---------------------------------------------------------------------------
// h1 = y2 @ u ; h2 = xt @ w2   (per-row dots)
// ---------------------------------------------------------------------------
__global__ __launch_bounds__(256) void k_h(const float* __restrict__ y2,
                                           const float* __restrict__ xtF,
                                           const float* __restrict__ u,
                                           const float* __restrict__ w2,
                                           float* __restrict__ h1,
                                           float* __restrict__ h2) {
  int i = blockIdx.x * 256 + threadIdx.x;        // 8192 threads
  float s1 = 0.f, s2 = 0.f;
#pragma unroll
  for (int k = 0; k < LAB_OUT; ++k) s1 += y2[(size_t)i * LAB_OUT + k] * u[k];
#pragma unroll
  for (int o = 0; o < OUT_F; ++o) s2 += xtF[(size_t)i * OUT_F + o] * w2[o];
  h1[i] = s1;
  h2[i] = s2;
}

// ---------------------------------------------------------------------------
// K4: z = softmax_j(tanh(h1_i + h2_j) + adj_ij) @ xt
// Flash-attention style: 16-row tile per block, 8 waves split columns,
// online softmax per wave, P(bf16) @ xt(bf16) via WMMA, LDS merge.
// ---------------------------------------------------------------------------
__global__ __launch_bounds__(256) void k_attn(const float* __restrict__ adj,
                                              const float* __restrict__ h1,
                                              const float* __restrict__ h2,
                                              const __bf16* __restrict__ xtB,
                                              float* __restrict__ z) {
  __shared__ float s_m[8 * 16], s_l[8 * 16], s_M[16], s_L[16];
  __shared__ float s_red[4 * 16 * OUT_F];        // 32 KB
  const int tid = threadIdx.x, w = tid >> 5, t = tid & 31;
  const int m16 = t & 15, g = t >> 4;
  const int row0 = blockIdx.x * 16;
  const float* arow = adj + (size_t)(row0 + m16) * N_ROWS;
  const float h1v = h1[row0 + m16];

  float rm = -INFINITY, rl = 0.f;
  v8f acc[8] = {};

  for (int ct = w; ct < N_ROWS / 32; ct += 8) {
    const int kb = ct * 32;
    __builtin_prefetch(arow + kb + 256, 0, 1);   // next tile for this wave
    alignas(16) float aj[16], hh[16];
    {
      const float* p0 = arow + kb + g * 8;
      const float* p1 = p0 + 16;
      *(float4*)(aj + 0)  = *(const float4*)(p0);
      *(float4*)(aj + 4)  = *(const float4*)(p0 + 4);
      *(float4*)(aj + 8)  = *(const float4*)(p1);
      *(float4*)(aj + 12) = *(const float4*)(p1 + 4);
      const float* q0 = h2 + kb + g * 8;
      const float* q1 = q0 + 16;
      *(float4*)(hh + 0)  = *(const float4*)(q0);
      *(float4*)(hh + 4)  = *(const float4*)(q0 + 4);
      *(float4*)(hh + 8)  = *(const float4*)(q1);
      *(float4*)(hh + 12) = *(const float4*)(q1 + 4);
    }
    float sv[16];
    float mx = -INFINITY;
#pragma unroll
    for (int j = 0; j < 16; ++j) {
      sv[j] = tanhf(h1v + hh[j]) + aj[j];
      mx = fmaxf(mx, sv[j]);
    }
    // row max across the lane pair holding the same row (t <-> t^16)
    mx = fmaxf(mx, __shfl_xor(mx, 16, 32));
    const float mnew = fmaxf(rm, mx);
    const float scale = __expf(rm - mnew);       // rm = -inf -> 0 first pass
    float ls = 0.f;
    v16bf pa;
#pragma unroll
    for (int j = 0; j < 16; ++j) {
      float p = __expf(sv[j] - mnew);
      ls += p;
      pa[j] = f2bf(p);
    }
    ls += __shfl_xor(ls, 16, 32);
    rl = rl * scale + ls;
    rm = mnew;
    // rescale accumulator (C-layout rows r+8g need the scale of row r+8g,
    // which lives in lane r+8g of the A-layout stats)
    float scr[8];
#pragma unroll
    for (int r = 0; r < 8; ++r) scr[r] = __shfl(scale, r + 8 * g, 32);
#pragma unroll
    for (int nc = 0; nc < 8; ++nc)
#pragma unroll
      for (int r = 0; r < 8; ++r) acc[nc][r] *= scr[r];
    // P @ xt tile (B tiles preloaded in groups of 4 for ILP)
    const __bf16* bp = xtB + ((size_t)ct * 8 * 32 + t) * 16;
#pragma unroll
    for (int h = 0; h < 2; ++h) {
      v16bf b0 = *(const v16bf*)(bp + (h * 4 + 0) * 512);
      v16bf b1 = *(const v16bf*)(bp + (h * 4 + 1) * 512);
      v16bf b2 = *(const v16bf*)(bp + (h * 4 + 2) * 512);
      v16bf b3 = *(const v16bf*)(bp + (h * 4 + 3) * 512);
      acc[h * 4 + 0] = wmma_bf16(pa, b0, acc[h * 4 + 0]);
      acc[h * 4 + 1] = wmma_bf16(pa, b1, acc[h * 4 + 1]);
      acc[h * 4 + 2] = wmma_bf16(pa, b2, acc[h * 4 + 2]);
      acc[h * 4 + 3] = wmma_bf16(pa, b3, acc[h * 4 + 3]);
    }
  }

  // --- merge 8 waves ---
  if (t < 16) { s_m[w * 16 + t] = rm; s_l[w * 16 + t] = rl; }
  __syncthreads();
  if (tid < 16) {
    float M = -INFINITY;
    for (int ww = 0; ww < 8; ++ww) M = fmaxf(M, s_m[ww * 16 + tid]);
    float L = 0.f;
    for (int ww = 0; ww < 8; ++ww)
      L += s_l[ww * 16 + tid] * __expf(s_m[ww * 16 + tid] - M);
    s_M[tid] = M;
    s_L[tid] = L;
  }
  __syncthreads();
  // rescale each wave's accumulator to the global max
#pragma unroll
  for (int r = 0; r < 8; ++r) {
    const int q = r + 8 * g;
    const float sc = __expf(s_m[w * 16 + q] - s_M[q]);
#pragma unroll
    for (int nc = 0; nc < 8; ++nc) acc[nc][r] *= sc;
  }
  __syncthreads();
  // deterministic tree reduction of accumulators across waves
  for (int step = 4; step >= 1; step >>= 1) {
    if (w >= step && w < 2 * step) {
      float* dst = s_red + (w - step) * (16 * OUT_F);
#pragma unroll
      for (int nc = 0; nc < 8; ++nc)
#pragma unroll
        for (int r = 0; r < 8; ++r)
          dst[(r + 8 * g) * OUT_F + nc * 16 + m16] = acc[nc][r];
    }
    __syncthreads();
    if (w < step) {
      const float* src = s_red + w * (16 * OUT_F);
#pragma unroll
      for (int nc = 0; nc < 8; ++nc)
#pragma unroll
        for (int r = 0; r < 8; ++r)
          acc[nc][r] += src[(r + 8 * g) * OUT_F + nc * 16 + m16];
    }
    __syncthreads();
  }
  if (w == 0) {
#pragma unroll
    for (int nc = 0; nc < 8; ++nc)
#pragma unroll
      for (int r = 0; r < 8; ++r) {
        const int q = r + 8 * g;
        z[(size_t)(row0 + q) * OUT_F + nc * 16 + m16] = acc[nc][r] / s_L[q];
      }
  }
}

// ---------------------------------------------------------------------------
extern "C" void kernel_launch(void* const* d_in, const int* in_sizes, int n_in,
                              void* d_out, int out_size, void* d_ws,
                              size_t ws_size, hipStream_t stream) {
  (void)in_sizes; (void)n_in; (void)out_size; (void)ws_size;
  const float* x   = (const float*)d_in[0];
  const float* adj = (const float*)d_in[1];
  const float* y   = (const float*)d_in[2];
  const float* gcn = (const float*)d_in[3];
  const float* Wt  = (const float*)d_in[4];
  const float* Wg  = (const float*)d_in[5];
  const float* Wtr = (const float*)d_in[6];
  const float* w1  = (const float*)d_in[7];
  const float* w2  = (const float*)d_in[8];

  float* z  = (float*)d_out;                               // [8192,128]
  float* y2 = (float*)d_out + (size_t)N_ROWS * OUT_F;      // [8192,64]

  char* ws = (char*)d_ws;
  __bf16* tmpB = (__bf16*)(ws + 0);          // 8192*64*2   = 1 MB
  __bf16* xtB  = (__bf16*)(ws + 1048576);    // 8192*128*2  = 2 MB
  __bf16* WtB  = (__bf16*)(ws + 3145728);    // 512*128*2   = 128 KB
  float*  xtF  = (float*) (ws + 3276800);    // 8192*128*4  = 4 MB
  float*  h1   = (float*) (ws + 7471104);    // 32 KB
  float*  h2   = (float*) (ws + 7503872);    // 32 KB
  float*  u    = (float*) (ws + 7536640);    // 256 B

  k_tmp   <<<(N_ROWS * LAB_OUT) / 256, 256, 0, stream>>>(y, Wg, tmpB);
  k_packWt<<<(IN_F * OUT_F) / 256,     256, 0, stream>>>(Wt, WtB);
  k_xt    <<<N_ROWS / 128,             256, 0, stream>>>(x, WtB, xtF);
  k_packXt<<<(N_ROWS * OUT_F) / 256,   256, 0, stream>>>(xtF, xtB);
  k_gcn   <<<N_ROWS / 16,              256, 0, stream>>>(gcn, tmpB, y2);
  k_u     <<<1, 64, 0, stream>>>(Wtr, w1, u);
  k_h     <<<N_ROWS / 256,             256, 0, stream>>>(y2, xtF, u, w2, h1, h2);
  k_attn  <<<N_ROWS / 16,              256, 0, stream>>>(adj, h1, h2, xtB, z);
}
